// ContrastiveLoss_38723425140992
// MI455X (gfx1250) — compile-verified, matrix operand order checked by reading, not run on hardware
//
#include <hip/hip_runtime.h>
#include <hip/hip_bf16.h>

// ---------------------------------------------------------------------------
// NT-Xent contrastive loss on MI455X (gfx1250, wave32).
//   K1: L2-normalize rows -> bf16 reps[8192][1024]   (16 MB, L2-resident)
//   K2: 128x128 sim tiles via V_WMMA_F32_16X16X32_BF16, K staged through LDS
//       with global_load_async_to_lds_b128 (ASYNCcnt) + double buffering.
//       Fused epilogue: exp(2*sim), diag masked, positives, row-sum atomics.
//   K3: loss = mean(log(denom) - 2*pos)
// ---------------------------------------------------------------------------

typedef __bf16 bf16_t;
typedef bf16_t v16bf __attribute__((ext_vector_type(16)));
typedef float  v8f   __attribute__((ext_vector_type(8)));
typedef unsigned short u16;
typedef unsigned int   u32;

#define DDIM   1024          // embedding dim
#define NB     4096          // batch B
#define TWOB   8192          // 2B rows
#define INV_T  2.0f          // 1 / TEMPERATURE

#define KC     32            // K chunk depth (one bf16 WMMA step)
#define NC     (DDIM / KC)   // 32 chunks
#define LROW   40            // LDS row stride in bf16 (32 data + 8 pad = 80B,
                             // 20 DWORDs: gcd(20,64)=4 -> 16 rows hit 16
                             // distinct bank groups, conflict-free b128 reads)

// float -> bf16, round-to-nearest-even
__device__ __forceinline__ u32 f2bf(float f) {
    u32 u = __float_as_uint(f);
    return (u + 0x7FFFu + ((u >> 16) & 1u)) >> 16;
}

// ---------------------------------------------------------------------------
// Kernel 1: L2-normalize each row of [emb_i; emb_j] into bf16 reps
// ---------------------------------------------------------------------------
__global__ void __launch_bounds__(256)
nt_xent_normalize(const float* __restrict__ emb_i,
                  const float* __restrict__ emb_j,
                  u16* __restrict__ reps) {
    __shared__ float red[8];
    const int row = blockIdx.x;
    const float* src = (row < NB) ? (emb_i + (size_t)row * DDIM)
                                  : (emb_j + (size_t)(row - NB) * DDIM);
    const int t = threadIdx.x;                 // 0..255, 4 floats each
    float4 v = reinterpret_cast<const float4*>(src)[t];
    float ss = v.x * v.x + v.y * v.y + v.z * v.z + v.w * v.w;

    #pragma unroll
    for (int off = 16; off > 0; off >>= 1) ss += __shfl_xor(ss, off, 32);
    const int lane = t & 31, wave = t >> 5;
    if (lane == 0) red[wave] = ss;
    __syncthreads();
    if (wave == 0) {
        float s = (lane < 8) ? red[lane] : 0.0f;
        #pragma unroll
        for (int off = 4; off > 0; off >>= 1) s += __shfl_xor(s, off, 32);
        if (lane == 0) red[0] = s;
    }
    __syncthreads();
    const float inv = 1.0f / fmaxf(sqrtf(red[0]), 1e-12f);

    uint2 packed;
    packed.x = (f2bf(v.y * inv) << 16) | f2bf(v.x * inv);
    packed.y = (f2bf(v.w * inv) << 16) | f2bf(v.z * inv);
    reinterpret_cast<uint2*>(reps + (size_t)row * DDIM)[t] = packed;
}

// ---------------------------------------------------------------------------
// CDNA5 async copy: global -> LDS, 16 bytes per lane, tracked by ASYNCcnt.
// ---------------------------------------------------------------------------
typedef __attribute__((address_space(3))) const u16 lds_cu16;

__device__ __forceinline__ unsigned lds_byte_addr(const void* p) {
    return (unsigned)(unsigned long long)(lds_cu16*)(const u16*)p;
}

__device__ __forceinline__ void async_copy_b128(unsigned lds_off, const void* gptr) {
    asm volatile("global_load_async_to_lds_b128 %0, %1, off"
                 :: "v"(lds_off), "v"(gptr) : "memory");
}

#define WAIT_ASYNC(n) asm volatile("s_wait_asynccnt " #n ::: "memory")

// Stage a 128-row x KC-col bf16 tile (rows gRowBase..+127, cols kc..kc+31)
// into LDS at byte offset ldsBase (row stride LROW*2 bytes). 256 threads,
// 2 async b128 instructions per thread (coalesced 16B chunks).
__device__ __forceinline__ void stage_tile(const u16* __restrict__ reps,
                                           int gRowBase, int kc,
                                           unsigned ldsBase, int t) {
    #pragma unroll
    for (int i = 0; i < 2; ++i) {
        const int item = t + 256 * i;          // 0..511
        const int row  = item >> 2;            // 0..127
        const int kq   = item & 3;             // 16B chunk within row
        const void* g  = reps + (size_t)(gRowBase + row) * DDIM + kc + kq * 8;
        async_copy_b128(ldsBase + row * (LROW * 2) + kq * 16, g);
    }
}

// ---------------------------------------------------------------------------
// Fragment fetch from an LDS tile, matching the WMMA 16x32 bf16 layout:
//   lanes 0-15 : row (lane&15), K = [0..7] and [16..23]
//   lanes16-31 : same rows,     K = [8..15] and [24..31]
// -> two ds_load_b128 at +0 / +32 bytes from the per-lane base.
// ---------------------------------------------------------------------------
union FragU { v16bf v; uint4 q[2]; };

__device__ __forceinline__ v16bf lds_frag(const u16* tile, int rowBase, int lane) {
    const u16* p = tile + (rowBase + (lane & 15)) * LROW + (lane >> 4) * 8;
    FragU f;
    f.q[0] = reinterpret_cast<const uint4*>(p)[0];
    f.q[1] = reinterpret_cast<const uint4*>(p)[2];
    return f.v;
}

#define WMMA_BF16(A, B, C) \
    __builtin_amdgcn_wmma_f32_16x16x32_bf16(false, (A), false, (B), (short)0, (C), false, false)

// ---------------------------------------------------------------------------
// Kernel 2: 128x128 tile of sim = reps . reps^T per 256-thread block.
// 8 waves in 4(M) x 2(N); each wave: 32 rows x 64 cols = 8 accumulators.
// ---------------------------------------------------------------------------
__global__ void __launch_bounds__(256)
nt_xent_sim_tile(const u16* __restrict__ reps,
                 float* __restrict__ row_denom,
                 float* __restrict__ row_pos) {
    __shared__ u16 shA[2][128 * LROW];         // 2 x 10 KB
    __shared__ u16 shB[2][128 * LROW];         // 2 x 10 KB   (40 KB total)

    const int t     = threadIdx.x;
    const int lane  = t & 31;
    const int wave  = t >> 5;
    const int waveM = wave & 3;                // 0..3 -> 32-row strip
    const int waveN = wave >> 2;               // 0..1 -> 64-col strip
    const int gM0   = blockIdx.y * 128;        // tile row base in sim
    const int gN0   = blockIdx.x * 128;        // tile col base in sim

    const unsigned ldsA[2] = { lds_byte_addr(&shA[0][0]), lds_byte_addr(&shA[1][0]) };
    const unsigned ldsB[2] = { lds_byte_addr(&shB[0][0]), lds_byte_addr(&shB[1][0]) };

    v8f acc[2][4];
    #pragma unroll
    for (int m = 0; m < 2; ++m)
        #pragma unroll
        for (int j = 0; j < 4; ++j) acc[m][j] = {};

    // prologue: stage chunk 0 into buffer 0  (4 async instructions/wave)
    stage_tile(reps, gM0, 0, ldsA[0], t);
    stage_tile(reps, gN0, 0, ldsB[0], t);

    for (int c = 0; c < NC; ++c) {
        const int cur = c & 1;
        if (c + 1 < NC) {
            // issue next chunk's copies, then wait only for the 4 older
            // (current-chunk) async instructions of this wave
            stage_tile(reps, gM0, (c + 1) * KC, ldsA[cur ^ 1], t);
            stage_tile(reps, gN0, (c + 1) * KC, ldsB[cur ^ 1], t);
            WAIT_ASYNC(4);
        } else {
            WAIT_ASYNC(0);
        }
        __syncthreads();                       // all waves' writes visible

        const u16* At = &shA[cur][0];
        const u16* Bt = &shB[cur][0];
        v16bf a0 = lds_frag(At, waveM * 32 + 0,  lane);
        v16bf a1 = lds_frag(At, waveM * 32 + 16, lane);
        v16bf b0 = lds_frag(Bt, waveN * 64 + 0,  lane);
        v16bf b1 = lds_frag(Bt, waveN * 64 + 16, lane);
        v16bf b2 = lds_frag(Bt, waveN * 64 + 32, lane);
        v16bf b3 = lds_frag(Bt, waveN * 64 + 48, lane);

        acc[0][0] = WMMA_BF16(a0, b0, acc[0][0]);
        acc[0][1] = WMMA_BF16(a0, b1, acc[0][1]);
        acc[0][2] = WMMA_BF16(a0, b2, acc[0][2]);
        acc[0][3] = WMMA_BF16(a0, b3, acc[0][3]);
        acc[1][0] = WMMA_BF16(a1, b0, acc[1][0]);
        acc[1][1] = WMMA_BF16(a1, b1, acc[1][1]);
        acc[1][2] = WMMA_BF16(a1, b2, acc[1][2]);
        acc[1][3] = WMMA_BF16(a1, b3, acc[1][3]);

        __syncthreads();                       // reads done before reuse
    }

    // Epilogue. C/D layout: VGPR r, lanes 0-15 -> M=+r, lanes 16-31 -> M=+8+r.
    const int khalf = lane >> 4;
    const int ncol  = lane & 15;
    #pragma unroll
    for (int m = 0; m < 2; ++m) {
        #pragma unroll
        for (int r = 0; r < 8; ++r) {
            const int gm = gM0 + waveM * 32 + m * 16 + r + 8 * khalf;
            float contrib = 0.0f;
            #pragma unroll
            for (int j = 0; j < 4; ++j) {
                const float s = acc[m][j][r];
                const int gn = gN0 + waveN * 64 + 16 * j + ncol;
                if (gn == (gm ^ NB)) row_pos[gm] = s;      // unique writer
                contrib += (gn == gm) ? 0.0f : __expf(s * INV_T);
            }
            // reduce the 16 lanes of one row (xor masks < 16 stay in-group)
            contrib += __shfl_xor(contrib, 8, 32);
            contrib += __shfl_xor(contrib, 4, 32);
            contrib += __shfl_xor(contrib, 2, 32);
            contrib += __shfl_xor(contrib, 1, 32);
            if (ncol == 0) atomicAdd(&row_denom[gm], contrib);
        }
    }
}

// ---------------------------------------------------------------------------
// Kernel 3: loss = mean over rows of (log(denom) - pos / T)
// ---------------------------------------------------------------------------
__global__ void __launch_bounds__(256)
nt_xent_finalize(const float* __restrict__ row_denom,
                 const float* __restrict__ row_pos,
                 float* __restrict__ out) {
    __shared__ float red[8];
    const int t = threadIdx.x;
    float sum = 0.0f;
    for (int r = t; r < TWOB; r += 256)
        sum += __logf(row_denom[r]) - row_pos[r] * INV_T;
    #pragma unroll
    for (int off = 16; off > 0; off >>= 1) sum += __shfl_xor(sum, off, 32);
    const int lane = t & 31, wave = t >> 5;
    if (lane == 0) red[wave] = sum;
    __syncthreads();
    if (t == 0) {
        float s = 0.0f;
        #pragma unroll
        for (int w = 0; w < 8; ++w) s += red[w];
        out[0] = s * (1.0f / (float)TWOB);
    }
}

// ---------------------------------------------------------------------------
extern "C" void kernel_launch(void* const* d_in, const int* in_sizes, int n_in,
                              void* d_out, int out_size, void* d_ws, size_t ws_size,
                              hipStream_t stream) {
    const float* emb_i = (const float*)d_in[0];
    const float* emb_j = (const float*)d_in[1];
    float* out = (float*)d_out;

    char* ws = (char*)d_ws;
    u16*   reps      = (u16*)ws;                                  // 16 MB bf16
    float* row_denom = (float*)(ws + (size_t)TWOB * DDIM * sizeof(u16));
    float* row_pos   = row_denom + TWOB;

    hipMemsetAsync(row_denom, 0, TWOB * sizeof(float), stream);

    nt_xent_normalize<<<TWOB, 256, 0, stream>>>(emb_i, emb_j, reps);

    dim3 grid(TWOB / 128, TWOB / 128);
    nt_xent_sim_tile<<<grid, 256, 0, stream>>>(reps, row_denom, row_pos);

    nt_xent_finalize<<<1, 256, 0, stream>>>(row_denom, row_pos, out);
}
